// AttentionBlock_7997229105196
// MI455X (gfx1250) — matrix operand order repaired and checked by reading, not compiled, not run on hardware
//
#include <hip/hip_runtime.h>
#include <hip/hip_bf16.h>
#include <stdint.h>

// Problem constants (from reference): B=2, T=4096, C=512, H=8, HD=64
#define BB 2
#define TT 4096
#define CC 512
#define NH 8
#define HD 64
#define N3C (3 * CC)        // 1536
#define SCALE 0.125f        // 1/sqrt(64)

typedef __attribute__((ext_vector_type(16))) __bf16 v16bf;
typedef __attribute__((ext_vector_type(8)))  float  v8f;
typedef __attribute__((ext_vector_type(4)))  unsigned int v4u;
typedef __attribute__((ext_vector_type(2)))  unsigned int v2u;
typedef __attribute__((ext_vector_type(4)))  float  v4f;

union FragBF {
  v16bf          v;
  v4u            q[2];     // two 16-byte halves -> global_load_b128 / ds_load_b128
  unsigned int   u32[8];
  unsigned short u16[16];
};

__device__ __forceinline__ unsigned short f2bf(float f) {
  __bf16 b = (__bf16)f;                       // native v_cvt to bf16
  return __builtin_bit_cast(unsigned short, b);
}
__device__ __forceinline__ unsigned int pack2bf(float a, float b) {
  return (unsigned int)f2bf(a) | ((unsigned int)f2bf(b) << 16);
}

// ---------------------------------------------------------------------------
// Prep A: convert x (fp32 [B*T, C]) -> bf16, same layout.  4 elems/thread.
// ---------------------------------------------------------------------------
__global__ void __launch_bounds__(256)
cvt_x_kernel(const float* __restrict__ x, unsigned short* __restrict__ xbf)
{
  const size_t i = ((size_t)blockIdx.x * 256 + threadIdx.x) * 4;
  v4f f = *(const v4f*)(x + i);
  v2u r;
  r.x = pack2bf(f.x, f.y);
  r.y = pack2bf(f.z, f.w);
  *(v2u*)(xbf + i) = r;
}

// ---------------------------------------------------------------------------
// Prep B: convert + transpose w_qkv (fp32 [C, 3C]) -> bf16 [3C, C]
// (K-contiguous for B-fragment b128 loads).  32x32 LDS tile.
// ---------------------------------------------------------------------------
__global__ void __launch_bounds__(256)
cvt_wt_kernel(const float* __restrict__ w, unsigned short* __restrict__ wt)
{
  __shared__ unsigned short tile[32][33];
  const int tx = threadIdx.x & 31;        // fast dim
  const int ty = threadIdx.x >> 5;        // 0..7
  const int nt = blockIdx.x % (N3C / 32); // column-tile
  const int kt = blockIdx.x / (N3C / 32); // row-tile
  const int k0 = kt * 32, n0 = nt * 32;
#pragma unroll
  for (int r = 0; r < 4; ++r) {
    const int k = k0 + ty + r * 8;
    tile[ty + r * 8][tx] = f2bf(w[(size_t)k * N3C + n0 + tx]);
  }
  __syncthreads();
#pragma unroll
  for (int r = 0; r < 4; ++r) {
    const int n = n0 + ty + r * 8;
    wt[(size_t)n * CC + k0 + tx] = tile[tx][ty + r * 8];
  }
}

// ---------------------------------------------------------------------------
// Kernel 1: qkv = x @ w_qkv + b_qkv  (bf16 WMMA, fp32 accumulate)
// One wave per 16x16 output tile.  Inner loop: 4x global_load_b128 + 1 WMMA.
// Writes Q (pre-scaled), K as [B,H,T,HD] bf16 and V^T as [B,H,HD,T] bf16.
// ---------------------------------------------------------------------------
__global__ void __launch_bounds__(256)
qkv_gemm_kernel(const unsigned short* __restrict__ xbf, // [B*T, C] bf16
                const unsigned short* __restrict__ wt,  // [3C, C]  bf16
                const float* __restrict__ bias,         // [3C] fp32
                unsigned short* __restrict__ qws,
                unsigned short* __restrict__ kws,
                unsigned short* __restrict__ vt)
{
  const int lane = threadIdx.x & 31;
  const int wave = threadIdx.x >> 5;
  const int tile = blockIdx.x * 8 + wave;
  const int ntn  = N3C / 16;                       // 96 column tiles
  const int m0   = (tile / ntn) * 16;
  const int n0   = (tile % ntn) * 16;
  const int hi   = (lane & 16) ? 1 : 0;

  // Per-lane base pointers (A: row-major x, consecutive-K; B: K-contiguous wt)
  const unsigned short* ap = xbf + (size_t)(m0 + (lane & 15)) * CC + hi * 8;
  const unsigned short* bp = wt  + (size_t)(n0 + (lane & 15)) * CC + hi * 8;

  v8f acc = {};
  for (int k0 = 0; k0 < CC; k0 += 32) {
    FragBF a, b;
    a.q[0] = *(const v4u*)(ap + k0);        // K = k0 + {0..7}  (+8 hi-half)
    a.q[1] = *(const v4u*)(ap + k0 + 16);   // K = k0 + 16 + {0..7}
    b.q[0] = *(const v4u*)(bp + k0);        // K = k0 + {0..7}  (+8 hi-half)
    b.q[1] = *(const v4u*)(bp + k0 + 16);
    acc = __builtin_amdgcn_wmma_f32_16x16x32_bf16(false, a.v, false, b.v,
                                                  (short)0, acc, false, false);
  }

  const int col  = n0 + (lane & 15);
  const float bv = bias[col];
  const int mat = col / CC;      // 0=Q 1=K 2=V
  const int cc  = col % CC;
  const int h   = cc / HD;
  const int d   = cc % HD;
#pragma unroll
  for (int r = 0; r < 8; ++r) {
    const int m  = m0 + r + hi * 8;          // C/D layout row
    const int bi = m / TT;
    const int t  = m % TT;
    const int bh = bi * NH + h;
    const float val = acc[r] + bv;
    if (mat == 0) {
      qws[((size_t)bh * TT + t) * HD + d] = f2bf(val * SCALE);
    } else if (mat == 1) {
      kws[((size_t)bh * TT + t) * HD + d] = f2bf(val);
    } else {
      vt[((size_t)bh * HD + d) * TT + t] = f2bf(val);   // transposed
    }
  }
}

// ---------------------------------------------------------------------------
// Kernel 2: causal flash attention.  8 waves/block; wave owns 16 query rows.
// All fragment loads are b128; per 32-key step: 8 WMMAs + online softmax.
// ---------------------------------------------------------------------------
__global__ void __launch_bounds__(256)
flash_attn_kernel(const unsigned short* __restrict__ qws,
                  const unsigned short* __restrict__ kws,
                  const unsigned short* __restrict__ vt,
                  float* __restrict__ out)        // [B, T, C] fp32
{
  __shared__ unsigned int lds_p[8][16 * 16];      // per wave: 16 rows x 32 keys bf16

  const int lane = threadIdx.x & 31;
  const int wave = threadIdx.x >> 5;
  const int qblk = blockIdx.x & 31;               // T/128 = 32 query blocks
  const int bh   = blockIdx.x >> 5;               // B*H = 16
  const int qw   = qblk * 128 + wave * 16;
  const int b    = bh >> 3;
  const int h    = bh & 7;
  const int ncol = lane & 15;
  const int hi   = (lane & 16) ? 1 : 0;

  const unsigned short* qb = qws + (size_t)bh * TT * HD;
  const unsigned short* kb = kws + (size_t)bh * TT * HD;
  const unsigned short* vb = vt  + (size_t)bh * HD * TT;

  // Resident Q A-fragments: two d-halves, each = 2x b128
  FragBF qa[2];
  {
    const unsigned short* qp = qb + (size_t)(qw + ncol) * HD + hi * 8;
#pragma unroll
    for (int hf = 0; hf < 2; ++hf) {
      qa[hf].q[0] = *(const v4u*)(qp + hf * 32);
      qa[hf].q[1] = *(const v4u*)(qp + hf * 32 + 16);
    }
  }

  v8f o[4] = {v8f{}, v8f{}, v8f{}, v8f{}};
  float Mrow[8], Lrow[8];
#pragma unroll
  for (int r = 0; r < 8; ++r) { Mrow[r] = -1e30f; Lrow[r] = 0.0f; }

  for (int key0 = 0; key0 <= qw + 15; key0 += 32) {
    // ---- S = Q K^T over 32 keys: two 16x16 f32 fragments ----
    v8f s0 = {}, s1 = {};
    {
      const unsigned short* ka = kb + (size_t)(key0 + ncol) * HD + hi * 8;
      const unsigned short* kc = ka + (size_t)16 * HD;   // keys +16
#pragma unroll
      for (int hf = 0; hf < 2; ++hf) {
        FragBF bk0, bk1;
        bk0.q[0] = *(const v4u*)(ka + hf * 32);
        bk0.q[1] = *(const v4u*)(ka + hf * 32 + 16);
        bk1.q[0] = *(const v4u*)(kc + hf * 32);
        bk1.q[1] = *(const v4u*)(kc + hf * 32 + 16);
        s0 = __builtin_amdgcn_wmma_f32_16x16x32_bf16(false, qa[hf].v, false, bk0.v,
                                                     (short)0, s0, false, false);
        s1 = __builtin_amdgcn_wmma_f32_16x16x32_bf16(false, qa[hf].v, false, bk1.v,
                                                     (short)0, s1, false, false);
      }
    }

    // ---- causal mask + online softmax (rows live in 16-lane halves) ----
    unsigned short* pp = (unsigned short*)lds_p[wave];
#pragma unroll
    for (int r = 0; r < 8; ++r) {
      const int m = r + hi * 8;
      const int q = qw + m;
      float v0 = (key0 + ncol      <= q) ? s0[r] : -1e30f;
      float v1 = (key0 + 16 + ncol <= q) ? s1[r] : -1e30f;
      float t = fmaxf(v0, v1);
#pragma unroll
      for (int off = 8; off >= 1; off >>= 1)
        t = fmaxf(t, __shfl_xor(t, off, 32));
      const float Mnew  = fmaxf(Mrow[r], t);
      const float alpha = __expf(Mrow[r] - Mnew);
      Mrow[r] = Mnew;
      const float p0 = __expf(v0 - Mnew);
      const float p1 = __expf(v1 - Mnew);
      float rs = p0 + p1;
#pragma unroll
      for (int off = 8; off >= 1; off >>= 1)
        rs += __shfl_xor(rs, off, 32);
      Lrow[r] = Lrow[r] * alpha + rs;
#pragma unroll
      for (int g = 0; g < 4; ++g) o[g][r] *= alpha;
      pp[m * 32 + ncol]      = f2bf(p0);
      pp[m * 32 + 16 + ncol] = f2bf(p1);
    }

    // ---- read P back in A-fragment layout: 2x ds_load_b128 ----
    FragBF pa;
    {
      const unsigned int* lp = lds_p[wave] + ncol * 16 + hi * 4;
      pa.q[0] = *(const v4u*)(lp);         // keys {0..7}  (+8 hi-half)
      pa.q[1] = *(const v4u*)(lp + 8);     // keys 16 + {0..7}
    }

    // ---- O += P * V  (V^T -> keys contiguous, 2x b128 per fragment) ----
#pragma unroll
    for (int g = 0; g < 4; ++g) {
      FragBF bv;
      const unsigned short* vp =
          vb + (size_t)(g * 16 + ncol) * TT + key0 + hi * 8;
      bv.q[0] = *(const v4u*)(vp);
      bv.q[1] = *(const v4u*)(vp + 16);
      o[g] = __builtin_amdgcn_wmma_f32_16x16x32_bf16(false, pa.v, false, bv.v,
                                                     (short)0, o[g], false, false);
    }
  }

  // ---- epilogue: divide by L, write fp32 [B,T,C] ----
#pragma unroll
  for (int r = 0; r < 8; ++r) {
    const int m = r + hi * 8;
    const int q = qw + m;
    const float inv = 1.0f / Lrow[r];
    float* op = out + ((size_t)(b * TT + q)) * CC + h * HD;
#pragma unroll
    for (int g = 0; g < 4; ++g)
      op[g * 16 + ncol] = o[g][r] * inv;
  }
}

// ---------------------------------------------------------------------------
extern "C" void kernel_launch(void* const* d_in, const int* in_sizes, int n_in,
                              void* d_out, int out_size, void* d_ws, size_t ws_size,
                              hipStream_t stream) {
  const float* x    = (const float*)d_in[0];   // [B,T,C]
  const float* w    = (const float*)d_in[1];   // [C,3C]
  const float* bias = (const float*)d_in[2];   // [3C]
  float*       out  = (float*)d_out;           // [B,T,C]

  // workspace (bf16): xbf 8MB | wt 1.5MB | Q 8MB | K 8MB | V^T 8MB  = 33.5 MB
  unsigned short* xbf = (unsigned short*)d_ws;
  unsigned short* wt  = xbf + (size_t)BB * TT * CC;
  unsigned short* qws = wt  + (size_t)CC * N3C;
  unsigned short* kws = qws + (size_t)BB * TT * CC;
  unsigned short* vtw = kws + (size_t)BB * TT * CC;

  const size_t nx = (size_t)BB * TT * CC;                  // 4M elements
  cvt_x_kernel<<<(int)(nx / (256 * 4)), 256, 0, stream>>>(x, xbf);
  cvt_wt_kernel<<<(CC / 32) * (N3C / 32), 256, 0, stream>>>(w, wt);

  const int tiles1 = (BB * TT / 16) * (N3C / 16);          // 49152 tiles
  qkv_gemm_kernel<<<tiles1 / 8, 256, 0, stream>>>(xbf, wt, bias, qws, kws, vtw);

  flash_attn_kernel<<<BB * NH * (TT / 128), 256, 0, stream>>>(qws, kws, vtw, out);
}
